// DynamicConv1x1_12704513262141
// MI455X (gfx1250) — matrix operand order repaired and checked by reading, not compiled, not run on hardware
//
#include <hip/hip_runtime.h>
#include <hip/hip_bf16.h>

// ---------------------------------------------------------------------------
// Dynamic 1x1 conv (CondConv): out[b] = (sum_e attn[b,e]*W[e]) @ x[b]
// B=32, C_in=C_out=256, E=8, HW=56*56=3136
// Main GEMM: v_wmma_f32_16x16x32_bf16, double-buffered LDS staging,
// A-tile staged via CDNA5 async global->LDS DMA when available.
// ---------------------------------------------------------------------------

typedef __bf16 bf16_t;
typedef __attribute__((ext_vector_type(16))) __bf16 v16bf;
typedef __attribute__((ext_vector_type(8)))  __bf16 v8bf;
typedef __attribute__((ext_vector_type(4)))  __bf16 v4bf;
typedef __attribute__((ext_vector_type(8)))  float  v8f;
typedef int v4i __attribute__((vector_size(4 * sizeof(int))));
typedef __attribute__((address_space(3))) v4i lds_v4i;

#define KB_B     32
#define KB_CIN   256
#define KB_COUT  256
#define KB_E     8
#define KB_CH    64          // C_in/4
#define KB_HW    3136        // 56*56
#define LDA      40          // 32 + 8 pad (bf16 elems) -> 80B row stride
#define LDB      40

#if defined(__has_builtin)
#  if __has_builtin(__builtin_amdgcn_global_load_async_to_lds_b128)
#    define HAVE_ASYNC_LDS 1
#  endif
#endif
#ifndef HAVE_ASYNC_LDS
#  define HAVE_ASYNC_LDS 0
#endif

__device__ __forceinline__ void async_wait0() {
#if HAVE_ASYNC_LDS
#  if __has_builtin(__builtin_amdgcn_s_wait_asynccnt)
    __builtin_amdgcn_s_wait_asynccnt(0);
#  else
    asm volatile("s_wait_asynccnt 0x0" ::: "memory");
#  endif
#endif
}

// ---------------------------------------------------------------------------
// Kernel 1: global average pool  pooled[b,c] = mean_hw x[b,c,:,:]
// ---------------------------------------------------------------------------
__global__ __launch_bounds__(256)
void pool_kernel(const float* __restrict__ x, float* __restrict__ pooled) {
    const int bc = blockIdx.x;                 // b*256 + c
    const float* p = x + (size_t)bc * KB_HW;
    float s = 0.f;
    for (int i = threadIdx.x; i < KB_HW; i += 256) s += p[i];
    __shared__ float red[256];
    red[threadIdx.x] = s;
    __syncthreads();
    for (int off = 128; off > 0; off >>= 1) {
        if (threadIdx.x < off) red[threadIdx.x] += red[threadIdx.x + off];
        __syncthreads();
    }
    if (threadIdx.x == 0) pooled[bc] = red[0] * (1.0f / KB_HW);
}

// ---------------------------------------------------------------------------
// Kernel 2: attention head. h=relu(pooled@w1^T+b1); logits=h@w2^T+b2; softmax.
// ---------------------------------------------------------------------------
__global__ __launch_bounds__(256)
void attn_kernel(const float* __restrict__ pooled,
                 const float* __restrict__ w1, const float* __restrict__ b1,
                 const float* __restrict__ w2, const float* __restrict__ b2,
                 float* __restrict__ attn) {
    __shared__ float hsh[KB_B * KB_CH];      // 32*64
    __shared__ float lsh[KB_B * KB_E];       // 32*8
    const int t = threadIdx.x;

    for (int i = t; i < KB_B * KB_CH; i += 256) {
        const int b = i >> 6, j = i & 63;
        const float* pb = pooled + b * KB_CIN;
        const float* wr = w1 + j * KB_CIN;
        float s = b1[j];
        #pragma unroll 4
        for (int k = 0; k < KB_CIN; ++k) s += pb[k] * wr[k];
        hsh[i] = s > 0.f ? s : 0.f;
    }
    __syncthreads();
    {
        const int b = t >> 3, e = t & 7;
        const float* hb = hsh + b * KB_CH;
        const float* wr = w2 + e * KB_CH;
        float s = b2[e];
        #pragma unroll 4
        for (int k = 0; k < KB_CH; ++k) s += hb[k] * wr[k];
        lsh[t] = s;
    }
    __syncthreads();
    if (t < KB_B) {
        float m = -1e30f;
        for (int e = 0; e < KB_E; ++e) m = fmaxf(m, lsh[t * KB_E + e]);
        float ex[KB_E], sum = 0.f;
        for (int e = 0; e < KB_E; ++e) { ex[e] = __expf(lsh[t * KB_E + e] - m); sum += ex[e]; }
        const float inv = 1.0f / sum;
        for (int e = 0; e < KB_E; ++e) attn[t * KB_E + e] = ex[e] * inv;
    }
}

// ---------------------------------------------------------------------------
// Kernel 3: w_dyn[b,o,c] = sum_e attn[b,e] * weights[e,o,c], written as bf16.
// ---------------------------------------------------------------------------
__global__ __launch_bounds__(256)
void wdyn_kernel(const float* __restrict__ weights,
                 const float* __restrict__ attn,
                 bf16_t* __restrict__ wdyn) {
    const int o = blockIdx.x, c = threadIdx.x;
    float we[KB_E];
    #pragma unroll
    for (int e = 0; e < KB_E; ++e)
        we[e] = weights[((size_t)e * KB_COUT + o) * KB_CIN + c];

    __shared__ float a[KB_B * KB_E];
    if (threadIdx.x < KB_B * KB_E) a[threadIdx.x] = attn[threadIdx.x];
    __syncthreads();

    for (int b = 0; b < KB_B; ++b) {
        float s = 0.f;
        #pragma unroll
        for (int e = 0; e < KB_E; ++e) s += a[b * KB_E + e] * we[e];
        wdyn[((size_t)b * KB_COUT + o) * KB_CIN + c] = (bf16_t)s;
    }
}

// ---------------------------------------------------------------------------
// Kernel 4: main per-sample GEMM via WMMA bf16, double-buffered staging.
//   out[b][m][n] = sum_k wdyn[b][m][k] * x[b][k][n]
// Block: 256 threads (8 waves). Block tile M=256, N=64, K-step 32.
// ---------------------------------------------------------------------------
__device__ __forceinline__ v16bf cat8(v8bf lo, v8bf hi) {
    v16bf r;
    #pragma unroll
    for (int i = 0; i < 8; ++i) { r[i] = lo[i]; r[i + 8] = hi[i]; }
    return r;
}

__global__ __launch_bounds__(256)
void dyn_gemm_kernel(const float* __restrict__ x,
                     const bf16_t* __restrict__ wdyn,
                     float* __restrict__ out) {
    const int b  = blockIdx.y;
    const int n0 = blockIdx.x * 64;

    __shared__ bf16_t lA[2][KB_COUT * LDA];   // [m][k] row-major, 20KB each
    __shared__ bf16_t lB[2][64 * LDB];        // [n][k] transposed x tile, 5KB each

    const float*  xb = x    + (size_t)b * KB_CIN  * KB_HW;
    const bf16_t* wb = wdyn + (size_t)b * KB_COUT * KB_CIN;

    const int tid  = threadIdx.x;
    const int lane = tid & 31;
    const int wave = tid >> 5;
    const int wm   = (wave & 3) * 64;      // wave M offset
    const int wn   = (wave >> 2) * 32;     // wave N offset

    const int lrow = lane & 15;
    const int lkoA = (lane >> 4) * 8;      // A: K base 0 / 8 (+16 for hi chunk)
    const int lkoB = (lane >> 4) * 8;      // B: K base 0 / 8 (+16 for hi chunk)

    // ---- staging helpers -------------------------------------------------
    // A tile: wdyn[b][0..255][k0..k0+31] -> lA[buf], 16KB, 4x 16B per thread
    auto stage_A = [&](int k0, int buf) {
        const int row0 = tid >> 2;             // 0..63
        const int kk   = (tid & 3) * 8;        // 0,8,16,24
        #pragma unroll
        for (int i = 0; i < 4; ++i) {
            const int row = row0 + i * 64;
#if HAVE_ASYNC_LDS
            __builtin_amdgcn_global_load_async_to_lds_b128(
                (v4i*)(wb + (size_t)row * KB_CIN + k0 + kk),
                (lds_v4i*)&lA[buf][row * LDA + kk], 0, 0);
#else
            *(uint4*)&lA[buf][row * LDA + kk] =
                *(const uint4*)&wb[(size_t)row * KB_CIN + k0 + kk];
#endif
        }
    };
    // B tile: x[b][k0..k0+31][n0..n0+63] f32, 8 values per thread into regs
    float breg[8];
    auto load_B = [&](int k0) {
        #pragma unroll
        for (int i = 0; i < 2; ++i) {
            const int idx = tid + i * 256;     // 0..511
            const int kg  = idx >> 6;          // 0..7 (group of 4 k)
            const int n   = idx & 63;
            #pragma unroll
            for (int j = 0; j < 4; ++j)
                breg[i * 4 + j] = xb[(size_t)(k0 + kg * 4 + j) * KB_HW + n0 + n];
        }
    };
    auto store_B = [&](int buf) {
        #pragma unroll
        for (int i = 0; i < 2; ++i) {
            const int idx = tid + i * 256;
            const int kg  = idx >> 6;
            const int n   = idx & 63;
            v4bf v;
            #pragma unroll
            for (int j = 0; j < 4; ++j) v[j] = (bf16_t)breg[i * 4 + j];
            *(v4bf*)&lB[buf][n * LDB + kg * 4] = v;   // 8B store, 8B aligned
        }
    };

    // ---- accumulators ----------------------------------------------------
    const v8f vzero = {0.f, 0.f, 0.f, 0.f, 0.f, 0.f, 0.f, 0.f};
    v8f acc[4][2];
    #pragma unroll
    for (int mt = 0; mt < 4; ++mt)
        #pragma unroll
        for (int nt = 0; nt < 2; ++nt) acc[mt][nt] = vzero;

    // ---- prologue: fill buffer 0 ----------------------------------------
    stage_A(0, 0);
    load_B(0);
    store_B(0);
    async_wait0();
    __syncthreads();

    // ---- main loop: 8 K-steps, ping-pong buffers -------------------------
    int cur = 0;
    for (int s = 0; s < 8; ++s) {
        const bool has_next = (s < 7);
        if (has_next) {
            stage_A((s + 1) * 32, cur ^ 1);    // async DMA into next buffer
            load_B((s + 1) * 32);              // global loads into registers
        }

        // compute from lA[cur], lB[cur]
        v16bf bfr[2];
        #pragma unroll
        for (int nt = 0; nt < 2; ++nt) {
            const int col = wn + nt * 16 + lrow;
            v8bf blo = *(const v8bf*)&lB[cur][col * LDB + lkoB];
            v8bf bhi = *(const v8bf*)&lB[cur][col * LDB + lkoB + 16];
            bfr[nt] = cat8(blo, bhi);
        }
        #pragma unroll
        for (int mt = 0; mt < 4; ++mt) {
            const int row = wm + mt * 16 + lrow;
            v8bf alo = *(const v8bf*)&lA[cur][row * LDA + lkoA];
            v8bf ahi = *(const v8bf*)&lA[cur][row * LDA + lkoA + 16];
            v16bf af = cat8(alo, ahi);
            #pragma unroll
            for (int nt = 0; nt < 2; ++nt) {
                acc[mt][nt] = __builtin_amdgcn_wmma_f32_16x16x32_bf16(
                    false, af, false, bfr[nt], (short)0, acc[mt][nt], false, false);
            }
        }

        if (has_next) {
            store_B(cur ^ 1);                  // convert + LDS store for next step
            async_wait0();                     // A DMA for next step complete
            __syncthreads();
            cur ^= 1;
        }
    }

    // ---- epilogue: C layout lane L -> N=L&15; VGPR r -> M = r + 8*(L>=16) -
    const int mhalf = (lane >> 4) * 8;
    #pragma unroll
    for (int mt = 0; mt < 4; ++mt) {
        #pragma unroll
        for (int nt = 0; nt < 2; ++nt) {
            const int mbase = wm + mt * 16 + mhalf;
            const int ncol  = n0 + wn + nt * 16 + (lane & 15);
            #pragma unroll
            for (int r = 0; r < 8; ++r) {
                out[((size_t)b * KB_COUT + mbase + r) * KB_HW + ncol] = acc[mt][nt][r];
            }
        }
    }
}

// ---------------------------------------------------------------------------
// Launch
// ---------------------------------------------------------------------------
extern "C" void kernel_launch(void* const* d_in, const int* in_sizes, int n_in,
                              void* d_out, int out_size, void* d_ws, size_t ws_size,
                              hipStream_t stream) {
    const float* x       = (const float*)d_in[0];   // [32,256,56,56]
    const float* weights = (const float*)d_in[1];   // [8,256,256]
    const float* w1      = (const float*)d_in[2];   // [64,256]
    const float* b1      = (const float*)d_in[3];   // [64]
    const float* w2      = (const float*)d_in[4];   // [8,64]
    const float* b2      = (const float*)d_in[5];   // [8]
    float* out = (float*)d_out;                     // [32,256,56,56]

    // workspace: pooled f32 [8192] | attn f32 [256] | (64KB align) wdyn bf16 [2M]
    float*  pooled = (float*)d_ws;
    float*  attn   = pooled + KB_B * KB_CIN;
    bf16_t* wdyn   = (bf16_t*)((char*)d_ws + (64 << 10));

    pool_kernel<<<KB_B * KB_CIN, 256, 0, stream>>>(x, pooled);
    attn_kernel<<<1, 256, 0, stream>>>(pooled, w1, b1, w2, b2, attn);
    wdyn_kernel<<<KB_COUT, 256, 0, stream>>>(weights, attn, wdyn);

    dim3 grid(KB_HW / 64, KB_B);   // 49 x 32
    dyn_gemm_kernel<<<grid, 256, 0, stream>>>(x, wdyn, out);
}
